// DDCD_Smooth_80805514706962
// MI455X (gfx1250) — matrix operand order repaired and checked by reading, not compiled, not run on hardware
//
#include <hip/hip_runtime.h>
#include <hip/hip_bf16.h>
#include <math.h>

typedef __attribute__((ext_vector_type(16))) _Float16 v16h;
typedef __attribute__((ext_vector_type(8)))  _Float16 v8h;
typedef __attribute__((ext_vector_type(8)))  float    v8f;

#define NT    8       // samples per workgroup
#define DNODE 256     // nodes
#define BW    15
#define HID   16
#define BT_STRIDE 264 // padded f16 row stride for sBt (132 dwords -> conflict-free)
#define SA_STRIDE 40  // padded f16 row stride for sA  (20 dwords  -> conflict-free)

// ---- native gfx1250 tanh if the toolchain exposes it -----------------
#if defined(__has_builtin)
#if __has_builtin(__builtin_amdgcn_tanhf)
#define TANHF(x) __builtin_amdgcn_tanhf((x))
#elif __has_builtin(__builtin_amdgcn_tanh_f32)
#define TANHF(x) __builtin_amdgcn_tanh_f32((x))
#endif
#endif
#ifndef TANHF
__device__ __forceinline__ float __tanh_fallback(float v) {
    return 1.0f - 2.0f / (__expf(2.0f * v) + 1.0f);
}
#define TANHF(x) __tanh_fallback((x))
#endif

// LDS weight-pool offsets (floats)
#define O_TW    0
#define O_TB    256
#define O_B0TW  272
#define O_B0TB  512
#define O_B0L1W 527
#define O_B0L1B 767
#define O_B0L2W 782
#define O_B0L2B 1007
#define O_B1TW  1022
#define O_B1TB  1262
#define O_B1L1W 1277
#define O_B1L1B 1517
#define O_B1L2W 1532
#define O_B1L2B 1757
#define O_FW1   1772
#define O_FB1   1997
#define O_FW2   2012
#define O_FB2   2027
#define W_TOTAL 2028

__global__ __launch_bounds__(256) void ddcd_fused_kernel(
    const float* __restrict__ x,     const float* __restrict__ t,
    const float* __restrict__ adjA,  const float* __restrict__ node_emb,
    const float* __restrict__ time_W,const float* __restrict__ time_b,
    const float* __restrict__ b0tW,  const float* __restrict__ b0tb,
    const float* __restrict__ b0l1W, const float* __restrict__ b0l1b,
    const float* __restrict__ b0l2W, const float* __restrict__ b0l2b,
    const float* __restrict__ b1tW,  const float* __restrict__ b1tb,
    const float* __restrict__ b1l1W, const float* __restrict__ b1l1b,
    const float* __restrict__ b1l2W, const float* __restrict__ b1l2b,
    const float* __restrict__ fW1,   const float* __restrict__ fb1,
    const float* __restrict__ fW2,   const float* __restrict__ fb2,
    float* __restrict__ out)
{
    // h_x transposed for WMMA B: Bt[col = n*16 + f][g], padded stride
    __shared__ __align__(16) _Float16 sBt[NT * 16 * BT_STRIDE];   // ~66 KB
    // adj^T K-slab: sA[h][gk], gk in [0,32), padded stride
    __shared__ __align__(16) _Float16 sA[DNODE * SA_STRIDE];      // 20 KB
    // h_z buffer: sHz[n][h][f(16)]
    __shared__ __align__(16) _Float16 sHz[NT * DNODE * 16];       // 64 KB
    __shared__ float sW[W_TOTAL];                                 //  8 KB
    __shared__ float sTe0[NT][BW];
    __shared__ float sTe1[NT][BW];

    const int tid = threadIdx.x;
    const int n0  = blockIdx.x * NT;

    // ---- Phase 0a: weights -> LDS ------------------------------------
#define CPW(off, src, cnt) \
    for (int i = tid; i < (cnt); i += 256) sW[(off) + i] = (src)[i];
    CPW(O_TW,    time_W, 256)  CPW(O_TB,    time_b, 16)
    CPW(O_B0TW,  b0tW,   240)  CPW(O_B0TB,  b0tb,   15)
    CPW(O_B0L1W, b0l1W,  240)  CPW(O_B0L1B, b0l1b,  15)
    CPW(O_B0L2W, b0l2W,  225)  CPW(O_B0L2B, b0l2b,  15)
    CPW(O_B1TW,  b1tW,   240)  CPW(O_B1TB,  b1tb,   15)
    CPW(O_B1L1W, b1l1W,  240)  CPW(O_B1L1B, b1l1b,  15)
    CPW(O_B1L2W, b1l2W,  225)  CPW(O_B1L2B, b1l2b,  15)
    CPW(O_FW1,   fW1,    225)  CPW(O_FB1,   fb1,    15)
    CPW(O_FW2,   fW2,    15)   CPW(O_FB2,   fb2,    1)
#undef CPW
    __syncthreads();

    // ---- Phase 0b: per-sample time embeddings (NT threads) -----------
    if (tid < NT) {
        const float tv = t[n0 + tid];
        float emb[16];
#pragma unroll
        for (int i = 0; i < 8; ++i) {
            const float fr  = __expf(-1.31576291028f * (float)i); // exp(-ln(1e4)/7*i)
            const float ang = tv * fr;
            emb[i]     = __sinf(ang);
            emb[i + 8] = __cosf(ang);
        }
        float ht[16];
#pragma unroll
        for (int j = 0; j < 16; ++j) {
            float s = sW[O_TB + j];
#pragma unroll
            for (int i = 0; i < 16; ++i) s += emb[i] * sW[O_TW + i * 16 + j];
            ht[j] = TANHF(s);
        }
#pragma unroll
        for (int j = 0; j < BW; ++j) {
            float s0 = sW[O_B0TB + j], s1 = sW[O_B1TB + j];
#pragma unroll
            for (int i = 0; i < 16; ++i) {
                s0 += ht[i] * sW[O_B0TW + i * BW + j];
                s1 += ht[i] * sW[O_B1TW + i * BW + j];
            }
            sTe0[tid][j] = TANHF(s0);
            sTe1[tid][j] = TANHF(s1);
        }
    }
    __syncthreads();

    // ---- Phase 1: per-(n,g) MLP stack -> sBt (f16, transposed) -------
    {
        const int g = tid;                       // node id fixed per thread
        float ne[BW];
#pragma unroll
        for (int j = 0; j < BW; ++j) ne[j] = TANHF(node_emb[g * BW + j]);
        float base0[BW];                         // block0-l1 partial (sample-invariant)
#pragma unroll
        for (int j = 0; j < BW; ++j) {
            float s = sW[O_B0L1B + j];
#pragma unroll
            for (int i = 0; i < BW; ++i) s += ne[i] * sW[O_B0L1W + (i + 1) * BW + j];
            base0[j] = s;
        }

#pragma unroll 1
        for (int n = 0; n < NT; ++n) {
            const float xv = x[(size_t)(n0 + n) * DNODE + g];
            const float tx = TANHF(xv);

            float h1[BW];
#pragma unroll
            for (int j = 0; j < BW; ++j)
                h1[j] = TANHF(base0[j] + tx * sW[O_B0L1W + j]) + sTe0[n][j];

            float h2[BW];
#pragma unroll
            for (int j = 0; j < BW; ++j) {
                float s = sW[O_B0L2B + j];
#pragma unroll
                for (int i = 0; i < BW; ++i) s += h1[i] * sW[O_B0L2W + i * BW + j];
                h2[j] = TANHF(s);
            }
            float w1[BW];                        // block1 l1, input=concat(h2, xv)
#pragma unroll
            for (int j = 0; j < BW; ++j) {
                float s = sW[O_B1L1B + j];
#pragma unroll
                for (int i = 0; i < BW; ++i) s += h2[i] * sW[O_B1L1W + i * BW + j];
                s += xv * sW[O_B1L1W + 15 * BW + j];
                w1[j] = TANHF(s) + sTe1[n][j];
            }
            float hx[BW];
#pragma unroll
            for (int j = 0; j < BW; ++j) {
                float s = sW[O_B1L2B + j];
#pragma unroll
                for (int i = 0; i < BW; ++i) s += w1[i] * sW[O_B1L2W + i * BW + j];
                hx[j] = TANHF(s);
            }
            const int colbase = n * 16;
#pragma unroll
            for (int f = 0; f < BW; ++f)
                sBt[(colbase + f) * BT_STRIDE + g] = (_Float16)hx[f];
            sBt[(colbase + 15) * BT_STRIDE + g] = (_Float16)0.0f;  // pad column
        }
    }

    // ---- Phase 2: adjacency einsum via WMMA --------------------------
    // C[h, col] = sum_g adjT[h,g] * Bt[col,g] ; adjT[h,g] = adjA[g*256+h]
    const int wv   = tid >> 5;           // wave id 0..7
    const int lane = tid & 31;
    const int lhi  = lane >> 4;          // lane half
    const int llo  = lane & 15;

    v8f acc[2][NT];
#pragma unroll
    for (int i = 0; i < 2; ++i)
#pragma unroll
        for (int c = 0; c < NT; ++c) acc[i][c] = (v8f)0.0f;

    for (int ks = 0; ks < 8; ++ks) {     // K slabs of 32 (K total = 256)
        __syncthreads();                  // protect sA reuse across slabs
        // stage adj^T slab: sA[h*SA_STRIDE + gk] = adjA[(ks*32+gk)*256 + h]
        for (int gk = 0; gk < 32; ++gk)
            sA[tid * SA_STRIDE + gk] =
                (_Float16)adjA[(size_t)(ks * 32 + gk) * DNODE + tid];
        __syncthreads();

        const int o1 = lhi << 4;         // A frag: K {0..15} | {16..31}
        // both A fragments (h-tiles wv*2 and wv*2+1) loaded once per slab
        v16h af0, af1;
        {
            const int h0 = (wv * 2 + 0) * 16 + llo;
            const int h1r = (wv * 2 + 1) * 16 + llo;
            *(v8h*)&af0       = *(const v8h*)&sA[h0 * SA_STRIDE + o1];
            *((v8h*)&af0 + 1) = *(const v8h*)&sA[h0 * SA_STRIDE + o1 + 8];
            *(v8h*)&af1       = *(const v8h*)&sA[h1r * SA_STRIDE + o1];
            *((v8h*)&af1 + 1) = *(const v8h*)&sA[h1r * SA_STRIDE + o1 + 8];
        }
        const int gb = ks * 32 + (lhi << 4);    // B frag K base per lane half
#pragma unroll
        for (int ct = 0; ct < NT; ++ct) {       // B frag loaded once, used twice
            const int col = ct * 16 + llo;      // B frag: N = llo
            v16h bf;
            *(v8h*)&bf       = *(const v8h*)&sBt[col * BT_STRIDE + gb];
            *((v8h*)&bf + 1) = *(const v8h*)&sBt[col * BT_STRIDE + gb + 8];
            acc[0][ct] = __builtin_amdgcn_wmma_f32_16x16x32_f16(
                false, af0, false, bf, (short)0, acc[0][ct], false, false);
            acc[1][ct] = __builtin_amdgcn_wmma_f32_16x16x32_f16(
                false, af1, false, bf, (short)0, acc[1][ct], false, false);
        }
    }

    // h_z = h_x - h_xw  -> sHz  (C layout: vgpr v, lane -> M=v+8*lhi, N=llo)
#pragma unroll
    for (int i = 0; i < 2; ++i) {
        const int htile = wv * 2 + i;
#pragma unroll
        for (int ct = 0; ct < NT; ++ct) {
            const int col = ct * 16 + llo;
            const int n   = col >> 4;
            const int f   = col & 15;
#pragma unroll
            for (int v = 0; v < 8; ++v) {
                const int h = htile * 16 + v + (lhi << 3);
                const float hxv = (float)sBt[col * BT_STRIDE + h];
                sHz[((n << 8) + h) * 16 + f] = (_Float16)(hxv - acc[i][ct][v]);
            }
        }
    }
    __syncthreads();

    // ---- Phase 3: final MLP 15->15->1 --------------------------------
#pragma unroll 1
    for (int n = 0; n < NT; ++n) {
        const int h = tid;
        const int base = ((n << 8) + h) * 16;
        float hz[BW];
#pragma unroll
        for (int f = 0; f < BW; ++f) hz[f] = (float)sHz[base + f];
        float z = sW[O_FB2];
#pragma unroll
        for (int j = 0; j < BW; ++j) {
            float s = sW[O_FB1 + j];
#pragma unroll
            for (int i = 0; i < BW; ++i) s += hz[i] * sW[O_FW1 + i * BW + j];
            z += TANHF(s) * sW[O_FW2 + j];
        }
        out[(size_t)(n0 + n) * DNODE + h] = z;
    }
}

extern "C" void kernel_launch(void* const* d_in, const int* in_sizes, int n_in,
                              void* d_out, int out_size, void* d_ws, size_t ws_size,
                              hipStream_t stream) {
    const float* x      = (const float*)d_in[0];
    const float* t      = (const float*)d_in[1];
    const float* adjA   = (const float*)d_in[2];
    const float* nemb   = (const float*)d_in[3];
    const float* time_W = (const float*)d_in[4];
    const float* time_b = (const float*)d_in[5];
    const float* b0tW   = (const float*)d_in[6];
    const float* b0tb   = (const float*)d_in[7];
    const float* b0l1W  = (const float*)d_in[8];
    const float* b0l1b  = (const float*)d_in[9];
    const float* b0l2W  = (const float*)d_in[10];
    const float* b0l2b  = (const float*)d_in[11];
    const float* b1tW   = (const float*)d_in[12];
    const float* b1tb   = (const float*)d_in[13];
    const float* b1l1W  = (const float*)d_in[14];
    const float* b1l1b  = (const float*)d_in[15];
    const float* b1l2W  = (const float*)d_in[16];
    const float* b1l2b  = (const float*)d_in[17];
    const float* fW1    = (const float*)d_in[18];
    const float* fb1    = (const float*)d_in[19];
    const float* fW2    = (const float*)d_in[20];
    const float* fb2    = (const float*)d_in[21];
    float* out = (float*)d_out;

    const int n = in_sizes[1];          // number of samples (t is [N])
    const int blocks = n / NT;
    ddcd_fused_kernel<<<blocks, 256, 0, stream>>>(
        x, t, adjA, nemb, time_W, time_b,
        b0tW, b0tb, b0l1W, b0l1b, b0l2W, b0l2b,
        b1tW, b1tb, b1l1W, b1l1b, b1l2W, b1l2b,
        fW1, fb1, fW2, fb2, out);
}
